// SkyFieldMS_17119739642228
// MI455X (gfx1250) — matrix operand order repaired and checked by reading, not compiled, not run on hardware
//
#include <hip/hip_runtime.h>
#include <math.h>

// ---------------------------------------------------------------------------
// SkyField MoE renderer for gfx1250 (MI455X).
// Route-first (top-1) + f16 WMMA MLP: 48->128->128->3 per ray.
// ---------------------------------------------------------------------------

typedef __attribute__((ext_vector_type(16))) _Float16 v16h;
typedef __attribute__((ext_vector_type(8)))  _Float16 v8h;
typedef __attribute__((ext_vector_type(8)))  float    v8f;

#define NEXP 8
#define HID  128
#define DIN  48
#define KP1  64            // layer-1 K padded to 64
#define W3N  16            // layer-3 N padded to 16
#define EXP_HALFS 26624    // 64*128 + 128*128 + 128*16 f16 per expert
#define OFF_W2 8192
#define OFF_W3 24576
#define ROWS_PER_BLOCK 128 // 8 waves * 16 rows

// ---------------- small pipeline kernels ----------------

__global__ void k_init(int* __restrict__ meta) {
    int t = threadIdx.x;
    if (t < 16) meta[t] = 0;   // counts[8], cursors[8]
}

__global__ void k_assign(const float* __restrict__ origins,
                         const float* __restrict__ centroids,
                         int* __restrict__ assign,
                         int* __restrict__ meta, int bs) {
    int i = blockIdx.x * blockDim.x + threadIdx.x;
    if (i >= bs) return;
    float ox = origins[i * 3 + 0];
    float oy = origins[i * 3 + 1];
    float oz = origins[i * 3 + 2];
    int best = 0;
    float bestd = 3.4e38f;
    #pragma unroll
    for (int e = 0; e < NEXP; ++e) {
        float dx = ox - centroids[e * 3 + 0];
        float dy = oy - centroids[e * 3 + 1];
        float dz = oz - centroids[e * 3 + 2];
        float d2 = dx * dx + dy * dy + dz * dz;
        if (d2 < bestd) { bestd = d2; best = e; }   // strict < keeps first (argmin tie rule)
    }
    assign[i] = best;
    atomicAdd(meta + best, 1);
}

__global__ void k_offsets(int* __restrict__ meta) {
    if (threadIdx.x != 0 || blockIdx.x != 0) return;
    int* counts = meta;
    int* ps     = meta + 16;
    int acc = 0;
    #pragma unroll
    for (int e = 0; e < NEXP; ++e) {
        ps[e] = acc;
        acc += (counts[e] + ROWS_PER_BLOCK - 1) & ~(ROWS_PER_BLOCK - 1);
    }
    ps[NEXP] = acc;
}

__global__ void k_scatter(const int* __restrict__ assign,
                          int* __restrict__ meta,
                          int* __restrict__ sorted, int bs) {
    int i = blockIdx.x * blockDim.x + threadIdx.x;
    if (i >= bs) return;
    int a = assign[i];
    int pos = meta[16 + a] + atomicAdd(meta + 8 + a, 1);
    sorted[pos] = i;
}

// Convert fp32 weights to f16, row-major, zero-padded.
__global__ void k_convert(const float* __restrict__ W1,
                          const float* __restrict__ W2,
                          const float* __restrict__ W3,
                          _Float16* __restrict__ wh) {
    int gid = blockIdx.x * blockDim.x + threadIdx.x;
    if (gid >= NEXP * EXP_HALFS) return;
    int e = gid / EXP_HALFS;
    int r = gid - e * EXP_HALFS;
    float v;
    if (r < OFF_W2) {                       // W1 padded: [64][128]
        int k = r >> 7, n = r & 127;
        v = (k < DIN) ? W1[(e * DIN + k) * HID + n] : 0.0f;
    } else if (r < OFF_W3) {                // W2: [128][128]
        int rr = r - OFF_W2;
        int k = rr >> 7, n = rr & 127;
        v = W2[(e * HID + k) * HID + n];
    } else {                                // W3 padded: [128][16]
        int rr = r - OFF_W3;
        int k = rr >> 4, n = rr & 15;
        v = (n < 3) ? W3[(e * HID + k) * 3 + n] : 0.0f;
    }
    wh[gid] = (_Float16)v;
}

// ---------------- WMMA fragment helpers ----------------

// A operand (16x32 f16): lane = M (0..15), hi = lane>>4 selects K groups of 8.
// Row-major LDS tile [16][stride]: two contiguous 16B chunks per lane.
__device__ __forceinline__ v16h loadA(const _Float16* base, int stride, int kt, int lane) {
    const int m  = lane & 15;
    const int hi = lane >> 4;
    const _Float16* p = base + m * stride + kt * 32 + hi * 8;
    v8h lo = *(const v8h*)p;
    v8h hv = *(const v8h*)(p + 16);
    v16h r;
    #pragma unroll
    for (int i = 0; i < 8; ++i) { r[i] = lo[i]; r[i + 8] = hv[i]; }
    return r;
}

// B operand (32x16 f16): lane = N (0..15), hi = lane>>4 selects K groups of 8 (same as A).
// Row-major weights [K][ldn]: strided column reads per lane.
__device__ __forceinline__ v16h loadB(const _Float16* w, int ldn, int kt, int n0, int lane) {
    const int n  = lane & 15;
    const int hi = lane >> 4;
    const _Float16* p = w + (size_t)(kt * 32 + hi * 8) * ldn + n0 + n;
    v16h r;
    #pragma unroll
    for (int i = 0; i < 8; ++i) { r[i] = p[(size_t)i * ldn]; r[i + 8] = p[(size_t)(i + 16) * ldn]; }
    return r;
}

// ---------------- main MLP kernel ----------------

__global__ __launch_bounds__(256) void k_mlp(
    const float* __restrict__ dirs, const float* __restrict__ app,
    const float* __restrict__ b1, const float* __restrict__ b2,
    const float* __restrict__ b3,
    const _Float16* __restrict__ wh,
    const int* __restrict__ meta, const int* __restrict__ sorted,
    float* __restrict__ out) {
    __shared__ alignas(128) _Float16 stage[8][16 * 128];  // per-wave tile
    __shared__ int sIdx[8][16];

    const int tid  = threadIdx.x;
    const int wave = tid >> 5;
    const int lane = tid & 31;
    const int m    = lane & 15;
    const int hi   = lane >> 4;

    const int* counts = meta;
    const int* ps     = meta + 16;
    const int base = blockIdx.x * ROWS_PER_BLOCK;
    if (base >= ps[NEXP]) return;

    int e = 0;
    while (e < NEXP - 1 && base >= ps[e + 1]) ++e;
    const int cnt = counts[e];
    const int st  = ps[e];
    const int rowbase = base + wave * 16;

    // gather ray id for this wave's row m (both lane halves need it)
    const int p   = rowbase + m;
    const int rel = p - st;
    const int id  = (rel < cnt) ? sorted[p] : -1;
    if (hi == 0) sIdx[wave][m] = id;
    const int idS = (id >= 0) ? id : 0;

    _Float16* stg = &stage[wave][0];

    // ---- build feature tile: 16 rows x 64 cols (stride 64), f16 ----
    if (hi == 0) {
        float x = dirs[idS * 3 + 0], y = dirs[idS * 3 + 1], z = dirs[idS * 3 + 2];
        float xx = x * x, yy = y * y, zz = z * z;
        float c[16];
        c[0]  = 0.28209479177387814f;
        c[1]  = -0.48860251190291987f * y;
        c[2]  =  0.48860251190291987f * z;
        c[3]  = -0.48860251190291987f * x;
        c[4]  =  1.0925484305920792f * x * y;
        c[5]  = -1.0925484305920792f * y * z;
        c[6]  =  0.94617469575755997f * zz - 0.31539156525251999f;
        c[7]  = -1.0925484305920792f * x * z;
        c[8]  =  0.54627421529603959f * (xx - yy);
        c[9]  =  0.59004358992664352f * y * (-3.0f * xx + yy);
        c[10] =  2.8906114426405538f * x * y * z;
        c[11] =  0.45704579946446572f * y * (1.0f - 5.0f * zz);
        c[12] =  0.3731763325901154f * z * (5.0f * zz - 3.0f);
        c[13] =  0.45704579946446572f * x * (1.0f - 5.0f * zz);
        c[14] =  1.4453057213202769f * z * (xx - yy);
        c[15] =  0.59004358992664352f * x * (-xx + 3.0f * yy);
        #pragma unroll
        for (int k = 0; k < 16; ++k) stg[m * KP1 + k] = (_Float16)c[k];
        #pragma unroll
        for (int k = DIN; k < KP1; ++k) stg[m * KP1 + k] = (_Float16)0.0f;
    } else {
        #pragma unroll
        for (int j = 0; j < 32; ++j)
            stg[m * KP1 + 16 + j] = (_Float16)app[idS * 32 + j];
    }
    asm volatile("s_wait_dscnt 0" ::: "memory");   // wave-local LDS visibility

    const _Float16* w1 = wh + (size_t)e * EXP_HALFS;

    // ---- layer 1: [16x64] x [64x128] ----
    v16h a0 = loadA(stg, KP1, 0, lane);
    v16h a1 = loadA(stg, KP1, 1, lane);
    v8f acc[8];
    #pragma unroll
    for (int nt = 0; nt < 8; ++nt) {
        v8f c = {};
        v16h b = loadB(w1, HID, 0, nt * 16, lane);
        c = __builtin_amdgcn_wmma_f32_16x16x32_f16(false, a0, false, b, (short)0, c, false, false);
        b = loadB(w1, HID, 1, nt * 16, lane);
        c = __builtin_amdgcn_wmma_f32_16x16x32_f16(false, a1, false, b, (short)0, c, false, false);
        float bias = b1[e * HID + nt * 16 + m];
        #pragma unroll
        for (int r = 0; r < 8; ++r) { float v = c[r] + bias; c[r] = v > 0.0f ? v : 0.0f; }
        acc[nt] = c;
    }
    // D layout -> row-major h1 [16][128] in LDS
    #pragma unroll
    for (int nt = 0; nt < 8; ++nt)
        #pragma unroll
        for (int r = 0; r < 8; ++r)
            stg[(r + 8 * hi) * HID + nt * 16 + m] = (_Float16)acc[nt][r];
    asm volatile("s_wait_dscnt 0" ::: "memory");

    // ---- layer 2: [16x128] x [128x128] ----
    v16h a[4];
    #pragma unroll
    for (int kt = 0; kt < 4; ++kt) a[kt] = loadA(stg, HID, kt, lane);
    const _Float16* w2 = w1 + OFF_W2;
    #pragma unroll
    for (int nt = 0; nt < 8; ++nt) {
        v8f c = {};
        #pragma unroll
        for (int kt = 0; kt < 4; ++kt) {
            v16h b = loadB(w2, HID, kt, nt * 16, lane);
            c = __builtin_amdgcn_wmma_f32_16x16x32_f16(false, a[kt], false, b, (short)0, c, false, false);
        }
        float bias = b2[e * HID + nt * 16 + m];
        #pragma unroll
        for (int r = 0; r < 8; ++r) { float v = c[r] + bias; c[r] = v > 0.0f ? v : 0.0f; }
        acc[nt] = c;
    }
    #pragma unroll
    for (int nt = 0; nt < 8; ++nt)
        #pragma unroll
        for (int r = 0; r < 8; ++r)
            stg[(r + 8 * hi) * HID + nt * 16 + m] = (_Float16)acc[nt][r];
    asm volatile("s_wait_dscnt 0" ::: "memory");

    // ---- layer 3: [16x128] x [128x16] (cols 0..2 valid) ----
    #pragma unroll
    for (int kt = 0; kt < 4; ++kt) a[kt] = loadA(stg, HID, kt, lane);
    const _Float16* w3 = w1 + OFF_W3;
    v8f c = {};
    #pragma unroll
    for (int kt = 0; kt < 4; ++kt) {
        v16h b = loadB(w3, W3N, kt, 0, lane);
        c = __builtin_amdgcn_wmma_f32_16x16x32_f16(false, a[kt], false, b, (short)0, c, false, false);
    }
    if (m < 3) {
        float bias = b3[e * 3 + m];
        #pragma unroll
        for (int r = 0; r < 8; ++r) {
            int row = r + 8 * hi;
            int rid = sIdx[wave][row];
            if (rid >= 0) {
                float v = c[r] + bias;
                out[rid * 3 + m] = 1.0f / (1.0f + __expf(-v));
            }
        }
    }
}

// ---------------- launcher ----------------

extern "C" void kernel_launch(void* const* d_in, const int* in_sizes, int n_in,
                              void* d_out, int out_size, void* d_ws, size_t ws_size,
                              hipStream_t stream) {
    const float* origins   = (const float*)d_in[0];
    const float* dirs      = (const float*)d_in[1];
    const float* app       = (const float*)d_in[2];
    const float* centroids = (const float*)d_in[3];
    const float* W1        = (const float*)d_in[4];
    const float* b1        = (const float*)d_in[5];
    const float* W2        = (const float*)d_in[6];
    const float* b2        = (const float*)d_in[7];
    const float* W3        = (const float*)d_in[8];
    const float* b3        = (const float*)d_in[9];
    float* out = (float*)d_out;

    const int bs = in_sizes[0] / 3;

    char* ws = (char*)d_ws;
    int* meta        = (int*)ws;                                   // counts[8], cursors[8], ps[9]
    int* assign      = (int*)(ws + 128);
    int* sorted      = (int*)(ws + 128 + (size_t)bs * 4);
    _Float16* wh     = (_Float16*)(ws + 128 + (size_t)bs * 4 + ((size_t)bs + 1024) * 4);

    k_init<<<1, 32, 0, stream>>>(meta);
    k_assign<<<(bs + 255) / 256, 256, 0, stream>>>(origins, centroids, assign, meta, bs);
    k_offsets<<<1, 1, 0, stream>>>(meta);
    k_scatter<<<(bs + 255) / 256, 256, 0, stream>>>(assign, meta, sorted, bs);
    k_convert<<<(NEXP * EXP_HALFS + 255) / 256, 256, 0, stream>>>(W1, W2, W3, wh);

    const int mlpBlocks = (bs + ROWS_PER_BLOCK - 1) / ROWS_PER_BLOCK + NEXP;
    k_mlp<<<mlpBlocks, 256, 0, stream>>>(dirs, app, b1, b2, b3, wh, meta, sorted, out);
}
